// GlideAttention_84215718739982
// MI455X (gfx1250) — matrix operand order repaired and checked by reading, not compiled, not run on hardware
//
#include <hip/hip_runtime.h>
#include <hip/hip_bf16.h>
#include <stdint.h>

// ---------------------------------------------------------------------------
// Glide attention for MI455X (gfx1250, wave32, WMMA bf16 16x16x32,
// async global->LDS staging, double-buffered GEMM + attention)
// B=2, QLEN=2048, HID=2048, NH=16, NKV=4, HD=128, BLOCK=4
// ---------------------------------------------------------------------------

typedef __attribute__((ext_vector_type(16))) __bf16 v16bf;
typedef __attribute__((ext_vector_type(8)))  float  v8f;

union AFrag16 { v16bf v; uint4 q[2]; };

__device__ __forceinline__ unsigned short f2bf_bits(float f) {
  union { float f; unsigned u; } x; x.f = f;
  unsigned r = x.u + 0x7fffu + ((x.u >> 16) & 1u); // round-to-nearest-even
  return (unsigned short)(r >> 16);
}

// CDNA5 async copy: LDS[vdst] = MEM[vaddr], 16B per lane, tracked by ASYNCcnt
__device__ __forceinline__ void async_ld_b128(unsigned short* lds,
                                              const unsigned short* g) {
  unsigned l = (unsigned)(uintptr_t)lds;
  asm volatile("global_load_async_to_lds_b128 %0, %1, off"
               :: "v"(l), "v"(g) : "memory");
}
__device__ __forceinline__ void wait_async0() {
  asm volatile("s_wait_asynccnt 0" ::: "memory");
}

// --------------------------- f32 -> bf16 convert ---------------------------
__global__ void cvt_f32_bf16(const float* __restrict__ in,
                             unsigned short* __restrict__ out, int n) {
  int i = blockIdx.x * blockDim.x + threadIdx.x;
  int stride = gridDim.x * blockDim.x;
  for (; i < n; i += stride) out[i] = f2bf_bits(in[i]);
}

// ---------------- f32 (2048x2048) -> bf16 transposed (n-major) -------------
__global__ __launch_bounds__(256) void cvt_t_w(const float* __restrict__ in,
                                               unsigned short* __restrict__ out) {
  __shared__ float tile[32][33];
  const int tx = threadIdx.x & 31, ty = threadIdx.x >> 5; // 32x8
  const int n0 = blockIdx.x * 32, k0 = blockIdx.y * 32;
  #pragma unroll
  for (int j = 0; j < 4; ++j)
    tile[ty + 8 * j][tx] = in[(size_t)(k0 + ty + 8 * j) * 2048 + n0 + tx];
  __syncthreads();
  #pragma unroll
  for (int j = 0; j < 4; ++j)
    out[(size_t)(n0 + ty + 8 * j) * 2048 + k0 + tx] =
        f2bf_bits(tile[tx][ty + 8 * j]);
}

// -------- v_cache (b,q,kvh,d) f32 -> Vt (b,kvh,d,q) bf16 (transposed) ------
__global__ __launch_bounds__(256) void cvt_t_v(const float* __restrict__ in,
                                               unsigned short* __restrict__ out) {
  __shared__ float tile[32][33];
  const int tx = threadIdx.x & 31, ty = threadIdx.x >> 5;
  const int z = blockIdx.z;            // b*4 + kvh
  const int b = z >> 2, kvh = z & 3;
  const int q0 = blockIdx.x * 32, d0 = blockIdx.y * 32;
  #pragma unroll
  for (int j = 0; j < 4; ++j)
    tile[ty + 8 * j][tx] =
        in[((size_t)(b * 2048 + q0 + ty + 8 * j) * 4 + kvh) * 128 + d0 + tx];
  __syncthreads();
  #pragma unroll
  for (int j = 0; j < 4; ++j)
    out[((size_t)z * 128 + d0 + ty + 8 * j) * 2048 + q0 + tx] =
        f2bf_bits(tile[tx][ty + 8 * j]);
}

// ------------------------------- bf16 GEMM ---------------------------------
// C(MxN,f32) = A(MxK bf16 row-major) * Bt(NxK bf16, pre-transposed) [+bias]
// 256 threads = 8 waves (4x2); macro tile 128x128; wave tile 32x64 (8 WMMAs);
// async global->LDS staging, double-buffered.
#define GTM 128
#define GTN 128
#define GTK 32
#define GLDA 40
#define GLDB 40
#define ABUF (GTM * GLDA)
#define BBUF (GTN * GLDB)

__global__ __launch_bounds__(256) void gemm_bf16(
    const unsigned short* __restrict__ A, const unsigned short* __restrict__ Bt,
    const float* __restrict__ bias, float* __restrict__ C,
    int M, int N, int K)
{
  __shared__ alignas(16) unsigned short sA[2 * ABUF];
  __shared__ alignas(16) unsigned short sB[2 * BBUF];
  const int t = threadIdx.x;
  const int lane = t & 31;
  const int w  = t >> 5;
  const int wr = w >> 1;   // 0..3 -> 32 rows each
  const int wc = w & 1;    // 0..1 -> 64 cols each
  const int m0 = blockIdx.x * GTM;
  const int n0 = blockIdx.y * GTN;
  const int lhalf = lane >> 4;
  const int l15   = lane & 15;

  auto stage = [&](int k0, int buf) {
    unsigned short* dA = &sA[buf * ABUF];
    unsigned short* dB = &sB[buf * BBUF];
    const int r = t >> 1, c = (t & 1) * 16;
    { // A: 128x32, 16 contiguous bf16 per thread -> two async b128
      const unsigned short* ga = &A[(size_t)(m0 + r) * K + k0 + c];
      async_ld_b128(&dA[r * GLDA + c], ga);
      async_ld_b128(&dA[r * GLDA + c + 8], ga + 8);
    }
    { // Bt: 128x32, 16 contiguous bf16 per thread -> two async b128
      const unsigned short* gb = &Bt[(size_t)(n0 + r) * K + k0 + c];
      async_ld_b128(&dB[r * GLDB + c], gb);
      async_ld_b128(&dB[r * GLDB + c + 8], gb + 8);
    }
  };

  v8f acc[2][4];
  #pragma unroll
  for (int mi = 0; mi < 2; ++mi)
    #pragma unroll
    for (int ni = 0; ni < 4; ++ni) acc[mi][ni] = {};

  const int nk = K >> 5;
  stage(0, 0);
  for (int kt = 0; kt < nk; ++kt) {
    const int cur = kt & 1;
    wait_async0();
    __syncthreads();
    if (kt + 1 < nk) stage((kt + 1) * GTK, cur ^ 1);

    const unsigned short* bA = &sA[cur * ABUF];
    const unsigned short* bB = &sB[cur * BBUF];
    AFrag16 af[2], bf[4];
    #pragma unroll
    for (int mi = 0; mi < 2; ++mi) {
      int m  = 32 * wr + 16 * mi + l15;
      int kh = 8 * lhalf;
      af[mi].q[0] = *reinterpret_cast<const uint4*>(&bA[m * GLDA + kh]);
      af[mi].q[1] = *reinterpret_cast<const uint4*>(&bA[m * GLDA + kh + 16]);
    }
    #pragma unroll
    for (int ni = 0; ni < 4; ++ni) {
      int n  = 64 * wc + 16 * ni + l15;
      int kb = 8 * lhalf;
      bf[ni].q[0] = *reinterpret_cast<const uint4*>(&bB[n * GLDB + kb]);
      bf[ni].q[1] = *reinterpret_cast<const uint4*>(&bB[n * GLDB + kb + 16]);
    }
    #pragma unroll
    for (int mi = 0; mi < 2; ++mi)
      #pragma unroll
      for (int ni = 0; ni < 4; ++ni)
        acc[mi][ni] = __builtin_amdgcn_wmma_f32_16x16x32_bf16(
            false, af[mi].v, false, bf[ni].v, (short)0, acc[mi][ni],
            false, false);
  }

  #pragma unroll
  for (int mi = 0; mi < 2; ++mi)
    #pragma unroll
    for (int ni = 0; ni < 4; ++ni) {
      int col = n0 + 64 * wc + 16 * ni + l15;
      float bv = bias ? bias[col] : 0.0f;
      #pragma unroll
      for (int r = 0; r < 8; ++r) {
        int row = m0 + 32 * wr + 16 * mi + r + 8 * lhalf;
        C[(size_t)row * N + col] = acc[mi][ni][r] + bv;
      }
    }
}

// --------------------------------- RoPE ------------------------------------
// qraw: (b, q, NH*HD) f32 -> Qb: (b, h, q, d) bf16, pre-scaled by 1/sqrt(HD)
__global__ void rope_kernel(const float* __restrict__ qraw,
                            const float* __restrict__ cosb,
                            const float* __restrict__ sinb,
                            unsigned short* __restrict__ Qb)
{
  int idx = blockIdx.x * blockDim.x + threadIdx.x;
  const int total = 2 * 16 * 2048 * 128;
  if (idx >= total) return;
  int d = idx & 127;
  int q = (idx >> 7) & 2047;
  int h = (idx >> 18) & 15;
  int b = idx >> 22;
  size_t rbase = ((size_t)(b * 2048 + q)) * 2048 + h * 128;
  float x  = qraw[rbase + d];
  float x2 = qraw[rbase + ((d < 64) ? (d + 64) : (d - 64))];
  size_t cb = ((size_t)(b * 2048 + q)) * 128 + d;
  float rot = (d < 64) ? -x2 : x2;
  const float scale = 0.08838834764831845f; // 1/sqrt(128), folded into Q
  Qb[idx] = f2bf_bits(scale * (x * cosb[cb] + rot * sinb[cb]));
}

// --------------------------- glide flash attention -------------------------
#define LDK 136
#define LDV 40
#define LDP 40
#define KBUF (32 * LDK)
#define VBUF (128 * LDV)

__global__ __launch_bounds__(256) void glide_attn(
    const unsigned short* __restrict__ Qb, const unsigned short* __restrict__ Kb,
    const unsigned short* __restrict__ Vt, unsigned short* __restrict__ Out)
{
  __shared__ alignas(16) unsigned short sK[2 * KBUF];      // [kv][hd] x2
  __shared__ alignas(16) unsigned short sV[2 * VBUF];      // [hd][kv] x2
  __shared__ alignas(16) unsigned short sP[8 * 16 * LDP];  // per-wave [16][32]

  const int t = threadIdx.x;
  const int lane = t & 31;
  const int w = t >> 5;
  const int lhalf = lane >> 4;
  const int l15 = lane & 15;
  const int bh = blockIdx.x;      // b*NH + h
  const int b = bh >> 4;
  const int h = bh & 15;
  const int kvh = h >> 2;         // GQA: 4 q-heads per kv-head
  const int R0 = blockIdx.y * 128;

  // Q fragments: 16x128 as 4 A-layout chunks, register-resident
  AFrag16 qf[4];
  {
    int qrow = R0 + 16 * w + l15;
    size_t base = ((size_t)bh * 2048 + qrow) * 128;
    int kh = 8 * lhalf;
    #pragma unroll
    for (int c = 0; c < 4; ++c) {
      qf[c].q[0] = *reinterpret_cast<const uint4*>(&Qb[base + 32 * c + kh]);
      qf[c].q[1] = *reinterpret_cast<const uint4*>(&Qb[base + 32 * c + kh + 16]);
    }
  }

  auto stageKV = [&](int it, int buf) {
    const int c0 = it * 32;
    unsigned short* dK = &sK[buf * KBUF];
    unsigned short* dV = &sV[buf * VBUF];
    int kv  = t >> 3;
    int col = (t & 7) * 16;
    const unsigned short* gk =
        &Kb[((size_t)((b * 2048 + c0 + kv) * 4 + kvh)) * 128 + col];
    async_ld_b128(&dK[kv * LDK + col],     gk);
    async_ld_b128(&dK[kv * LDK + col + 8], gk + 8);
    int d   = t >> 1;
    int kvo = (t & 1) * 16;
    const unsigned short* gv =
        &Vt[((size_t)(b * 4 + kvh) * 128 + d) * 2048 + c0 + kvo];
    async_ld_b128(&dV[d * LDV + kvo],     gv);
    async_ld_b128(&dV[d * LDV + kvo + 8], gv + 8);
  };

  v8f acc[8];
  #pragma unroll
  for (int n = 0; n < 8; ++n) acc[n] = {};
  float mrow[8], lrow[8];
  #pragma unroll
  for (int r = 0; r < 8; ++r) { mrow[r] = -1e30f; lrow[r] = 0.0f; }

  const int niter = (R0 + 128) >> 5;        // uniform across block
  stageKV(0, 0);
  for (int it = 0; it < niter; ++it) {
    const int cur = it & 1;
    const int c0 = it * 32;
    wait_async0();
    __syncthreads();
    if (it + 1 < niter) stageKV(it + 1, cur ^ 1);
    const unsigned short* bK = &sK[cur * KBUF];
    const unsigned short* bV = &sV[cur * VBUF];

    // S(16x32) = Q * K_tile^T : two 16x16 WMMA accumulators, 4-step K loop
    v8f s[2];
    #pragma unroll
    for (int sub = 0; sub < 2; ++sub) {
      v8f sacc = {};
      int kvr = sub * 16 + l15;
      #pragma unroll
      for (int c = 0; c < 4; ++c) {
        AFrag16 bfr;
        int hd0 = 32 * c + 8 * lhalf;
        bfr.q[0] = *reinterpret_cast<const uint4*>(&bK[kvr * LDK + hd0]);
        bfr.q[1] = *reinterpret_cast<const uint4*>(&bK[kvr * LDK + hd0 + 16]);
        sacc = __builtin_amdgcn_wmma_f32_16x16x32_bf16(false, qf[c].v, false,
                                                       bfr.v, (short)0, sacc,
                                                       false, false);
      }
      s[sub] = sacc;
    }

    // glide mask + online softmax (row state aligned with C layout)
    int j0 = c0 + l15;
    int j1 = c0 + 16 + l15;
    #pragma unroll
    for (int r = 0; r < 8; ++r) {
      int irow = R0 + 16 * w + r + 8 * lhalf;
      float a0 = s[0][r], a1 = s[1][r];
      bool vis0 = ((j0 >> 2) < (irow >> 2)) || (irow < 4 && j0 < 4);
      bool vis1 = ((j1 >> 2) < (irow >> 2)) || (irow < 4 && j1 < 4);
      a0 = vis0 ? a0 : -1e30f;    // scale pre-folded into Q
      a1 = vis1 ? a1 : -1e30f;
      float mx = fmaxf(a0, a1);
      for (int sh = 1; sh < 16; sh <<= 1) mx = fmaxf(mx, __shfl_xor(mx, sh));
      float mnew = fmaxf(mrow[r], mx);
      float sc = __expf(mrow[r] - mnew);
      float p0 = __expf(a0 - mnew);
      float p1 = __expf(a1 - mnew);
      float ps = p0 + p1;
      for (int sh = 1; sh < 16; sh <<= 1) ps += __shfl_xor(ps, sh);
      lrow[r] = lrow[r] * sc + ps;
      mrow[r] = mnew;
      #pragma unroll
      for (int n = 0; n < 8; ++n) acc[n][r] *= sc;
      int prow = w * 16 + r + 8 * lhalf;   // P: C-layout -> LDS (A-layout src)
      sP[prow * LDP + l15]      = f2bf_bits(p0);
      sP[prow * LDP + 16 + l15] = f2bf_bits(p1);
    }
    __syncthreads();

    // acc(16x128) += P(16x32) * V(32x128): 8 WMMAs sharing one P A-frag
    AFrag16 pf;
    {
      int prow = w * 16 + l15;
      int kh = 8 * lhalf;
      pf.q[0] = *reinterpret_cast<const uint4*>(&sP[prow * LDP + kh]);
      pf.q[1] = *reinterpret_cast<const uint4*>(&sP[prow * LDP + kh + 16]);
    }
    #pragma unroll
    for (int n = 0; n < 8; ++n) {
      AFrag16 vf;
      int vrow = 16 * n + l15;
      int kb = 8 * lhalf;
      vf.q[0] = *reinterpret_cast<const uint4*>(&bV[vrow * LDV + kb]);
      vf.q[1] = *reinterpret_cast<const uint4*>(&bV[vrow * LDV + kb + 16]);
      acc[n] = __builtin_amdgcn_wmma_f32_16x16x32_bf16(false, pf.v, false,
                                                       vf.v, (short)0, acc[n],
                                                       false, false);
    }
  }

  // epilogue: out = acc / l, bf16, layout (b, q, NH*HD)
  #pragma unroll
  for (int r = 0; r < 8; ++r) {
    float inv = 1.0f / lrow[r];
    int qrow = R0 + 16 * w + r + 8 * lhalf;
    size_t rb = ((size_t)(b * 2048 + qrow)) * 2048 + h * 128;
    #pragma unroll
    for (int n = 0; n < 8; ++n)
      Out[rb + 16 * n + l15] = f2bf_bits(acc[n][r] * inv);
  }
}

// ------------------------------ host launcher ------------------------------
extern "C" void kernel_launch(void* const* d_in, const int* in_sizes, int n_in,
                              void* d_out, int out_size, void* d_ws, size_t ws_size,
                              hipStream_t stream)
{
  (void)in_sizes; (void)n_in; (void)out_size; (void)ws_size;
  const float* hidden = (const float*)d_in[0];
  const float* kc     = (const float*)d_in[1];
  const float* vc     = (const float*)d_in[2];
  const float* cosb   = (const float*)d_in[3];
  const float* sinb   = (const float*)d_in[4];
  const float* Wq     = (const float*)d_in[5];
  const float* bq     = (const float*)d_in[6];
  const float* Wo     = (const float*)d_in[7];
  float* out = (float*)d_out;

  char* ws = (char*)d_ws;
  unsigned short* Xb   = (unsigned short*)(ws);                  // 16 MiB
  unsigned short* Wqt  = (unsigned short*)(ws + (16ull << 20));  //  8 MiB (N x K)
  unsigned short* Wot  = (unsigned short*)(ws + (24ull << 20));  //  8 MiB (N x K)
  unsigned short* Kb   = (unsigned short*)(ws + (32ull << 20));  //  4 MiB
  unsigned short* Vt   = (unsigned short*)(ws + (36ull << 20));  //  4 MiB (b,kvh,d,q)
  unsigned short* Qb   = (unsigned short*)(ws + (40ull << 20));  // 16 MiB
  float*          qraw = (float*)(ws + (56ull << 20));           // 32 MiB
  unsigned short* Attn = (unsigned short*)(ws + (56ull << 20));  // alias qraw

  cvt_f32_bf16<<<8192, 256, 0, stream>>>(hidden, Xb, 2 * 2048 * 2048);
  cvt_f32_bf16<<<2048, 256, 0, stream>>>(kc, Kb, 2 * 2048 * 4 * 128);

  dim3 gt(64, 64);
  cvt_t_w<<<gt, 256, 0, stream>>>(Wq, Wqt);
  cvt_t_w<<<gt, 256, 0, stream>>>(Wo, Wot);
  dim3 gv(64, 4, 8);
  cvt_t_v<<<gv, 256, 0, stream>>>(vc, Vt);

  dim3 gg(32, 16); // (M/128, N/128)
  gemm_bf16<<<gg, 256, 0, stream>>>(Xb, Wqt, bq, qraw, 4096, 2048, 2048);

  rope_kernel<<<32768, 256, 0, stream>>>(qraw, cosb, sinb, Qb);

  dim3 ga(32, 16); // (B*NH, QLEN/128)
  glide_attn<<<ga, 256, 0, stream>>>(Qb, Kb, Vt, Attn);

  gemm_bf16<<<gg, 256, 0, stream>>>(Attn, Wot, nullptr, out, 4096, 2048, 2048);
}
